// GritAdapterLayer_4114578669939
// MI455X (gfx1250) — matrix operand order repaired and checked by reading, not compiled, not run on hardware
//
#include <hip/hip_runtime.h>
#include <hip/hip_bf16.h>
#include <stdint.h>

// ---------------- CDNA5 WMMA / TDM types ----------------
typedef __attribute__((ext_vector_type(16))) __bf16 v16bf;
typedef __attribute__((ext_vector_type(8)))  __bf16 v8bf;
typedef __attribute__((ext_vector_type(8)))  float  v8f;
typedef __attribute__((ext_vector_type(4)))  unsigned v4u;
typedef __attribute__((ext_vector_type(8)))  int     v8i;
typedef __attribute__((ext_vector_type(4)))  int     v4i;
typedef unsigned short bfr;   // raw bf16 storage

#define N_NODES 8192
#define SEQ     4
#define DIMV    512
#define NHEAD   4
#define DHEAD   128
#define E_TOT   73728
#define BOUT    128
#define NLAY    2

// GEMM macro-tile: 8 waves x (32 rows x 64 cols) each
#define BM 256
#define BN 64
#define KC 64

// ---------------- helpers ----------------
__device__ __forceinline__ bfr f2bf(float f) {
  unsigned u = __float_as_uint(f);
  unsigned r = (u + 0x7FFFu + ((u >> 16) & 1u)) >> 16;  // RNE
  return (bfr)r;
}

// A fragment: 16x32 bf16, lane = (khalf<<4)|m ; elems: k = khalf*16 + e
__device__ __forceinline__ v16bf ldA(const bfr* __restrict__ row, int k0, int kh) {
  v8bf lo = *reinterpret_cast<const v8bf*>(row + k0 + kh * 16);
  v8bf hi = *reinterpret_cast<const v8bf*>(row + k0 + kh * 16 + 8);
  v16bf r;
#pragma unroll
  for (int i = 0; i < 8; i++) { r[i] = lo[i]; r[i + 8] = hi[i]; }
  return r;
}
// B fragment from B^T (N-major, K contiguous): lane n holds k = khalf*16 + 0..15
__device__ __forceinline__ v16bf ldB(const bfr* __restrict__ row, int k0, int kh) {
  v8bf lo = *reinterpret_cast<const v8bf*>(row + k0 + kh * 16);
  v8bf hi = *reinterpret_cast<const v8bf*>(row + k0 + kh * 16 + 8);
  v16bf r;
#pragma unroll
  for (int i = 0; i < 8; i++) { r[i] = lo[i]; r[i + 8] = hi[i]; }
  return r;
}
__device__ __forceinline__ v8f wmma_bf16(v16bf a, v16bf b, v8f c) {
  return __builtin_amdgcn_wmma_f32_16x16x32_bf16(false, a, false, b, (short)0, c,
                                                 false, false);
}

// ---------------- TDM: 2-D tile load Global -> LDS ----------------
// D# per CDNA5 ISA ch.8: group0 = {ctrl, lds_addr, global_addr, type=2},
// group1 = {data_size=2B, tensor dims, tile dims, dim0 stride}. 2-D: groups 2/3 zero.
__device__ __forceinline__ void tdm_load_2d(unsigned lds_off, const void* gaddr,
                                            unsigned tensor_d0, unsigned tensor_d1,
                                            unsigned tile_d0, unsigned tile_d1,
                                            unsigned long long stride_e) {
  unsigned long long ga = (unsigned long long)(uintptr_t)gaddr;
  v4u g0;
  g0[0] = 1u;                                           // count=1, user mode
  g0[1] = lds_off;                                      // LDS byte address
  g0[2] = (unsigned)ga;                                 // global_addr[31:0]
  g0[3] = (unsigned)((ga >> 32) & 0x01FFFFFFu) | (2u << 30);  // [56:32] | type=2
  v8i g1;
  g1[0] = (int)(1u << 16);                              // data_size code 1 (2B)
  g1[1] = (int)((tensor_d0 & 0xFFFFu) << 16);           // tensor_dim0[15:0] @bit48
  g1[2] = (int)(((tensor_d0 >> 16) & 0xFFFFu) | ((tensor_d1 & 0xFFFFu) << 16));
  g1[3] = (int)(((tensor_d1 >> 16) & 0xFFFFu) | ((tile_d0 & 0xFFFFu) << 16));
  g1[4] = (int)(tile_d1 & 0xFFFFu);                     // tile_dim1, tile_dim2=0
  g1[5] = (int)(unsigned)(stride_e & 0xFFFFFFFFull);    // dim0 stride[31:0]
  g1[6] = (int)(unsigned)((stride_e >> 32) & 0xFFFFull);// stride[47:32], dim1_stride=0
  g1[7] = 0;
  v4i z = {0, 0, 0, 0};
#if defined(__clang_major__) && (__clang_major__ >= 23)
  v8i z8 = {0, 0, 0, 0, 0, 0, 0, 0};
  __builtin_amdgcn_tensor_load_to_lds(g0, g1, z, z, z8, 0);
#else
  __builtin_amdgcn_tensor_load_to_lds(g0, g1, z, z, 0);
#endif
}

// ---------------- weight convert + transpose (f32 KxN -> bf16 NxK) ----------------
__global__ void k_wconv(const float* __restrict__ W, bfr* __restrict__ Wt,
                        int K, int Nc) {
  int t = blockIdx.x * blockDim.x + threadIdx.x;
  if (t >= K * Nc) return;
  int k = t / Nc, n = t % Nc;
  Wt[(size_t)n * K + k] = f2bf(W[t]);
}

// ---------------- self-loop scatter ----------------
__global__ void k_selfloop(const int* __restrict__ src, const int* __restrict__ dst,
                           const float* __restrict__ emb, float* __restrict__ nraw,
                           int E) {
  int e = blockIdx.x * blockDim.x + threadIdx.x;
  if (e >= E) return;
  int s = src[e];
  if (s == dst[e]) {
#pragma unroll
    for (int j = 0; j < 8; j++) atomicAdd(&nraw[(size_t)s * 8 + j], emb[(size_t)e * 8 + j]);
  }
}

// ---------------- first MLP layer (K=8, VALU) -> relu -> bf16 ----------------
__global__ void k_mlp1(const float* __restrict__ X, const float* __restrict__ W1,
                       const float* __restrict__ b1, bfr* __restrict__ H1, int M) {
  int t = blockIdx.x * blockDim.x + threadIdx.x;
  if (t >= M * DIMV) return;
  int m = t >> 9, c = t & 511;
  float acc = b1[c];
#pragma unroll
  for (int j = 0; j < 8; j++) acc = fmaf(X[(size_t)m * 8 + j], W1[j * DIMV + c], acc);
  H1[t] = f2bf(fmaxf(acc, 0.f));
}

// ---------------- TDM + LDS double-buffered bf16 WMMA GEMM -------------------
// C(MxN) = A(MxK) @ B(KxN). A row-major bf16; Bt = B^T (Nc x K) bf16.
// 8 waves; macro-tile BM(256) x BN(64); wave w -> rows [w*32, w*32+32) x BN cols.
// Wave 0 drives the Tensor Data Mover; everyone computes from LDS.
// Per k-step: 12 ds_load_b128 feed 8 v_wmma (A reuse x4, B reuse x2).
// Requires M%256==0, Nc%64==0, K%64==0 (true for all call sites).
__global__ __launch_bounds__(256) void k_gemm(
    const bfr* __restrict__ A, const bfr* __restrict__ Bt,
    const float* __restrict__ bias, float* __restrict__ Cf, bfr* __restrict__ Cb,
    int M, int K, int Nc, int relu) {
  __shared__ __align__(16) bfr Asm[2][BM * KC];
  __shared__ __align__(16) bfr Bsm[2][BN * KC];
  int w = threadIdx.x >> 5;
  int lane = threadIdx.x & 31;
  int kh = lane >> 4, nl = lane & 15;
  int bm = blockIdx.x % (M >> 8);
  int bn = blockIdx.x / (M >> 8);
  const bfr* Ablk = A + (size_t)bm * BM * K;
  const bfr* Bblk = Bt + (size_t)bn * BN * K;
  int nk = K / KC;
  bool issuer = (w == 0);                 // wave-uniform
  if (issuer) {
    tdm_load_2d((unsigned)(uintptr_t)&Asm[0][0], Ablk, K, BM, KC, BM, K);
    tdm_load_2d((unsigned)(uintptr_t)&Bsm[0][0], Bblk, K, BN, KC, BN, K);
  }
  v8f acc[2][4];
#pragma unroll
  for (int g = 0; g < 2; g++)
#pragma unroll
    for (int ct = 0; ct < 4; ct++)
#pragma unroll
      for (int r = 0; r < 8; r++) acc[g][ct][r] = 0.f;

  for (int i = 0; i < nk; i++) {
    int buf = i & 1;
    if (issuer) {
      if (i + 1 < nk) {
        tdm_load_2d((unsigned)(uintptr_t)&Asm[buf ^ 1][0],
                    Ablk + (size_t)(i + 1) * KC, K, BM, KC, BM, K);
        tdm_load_2d((unsigned)(uintptr_t)&Bsm[buf ^ 1][0],
                    Bblk + (size_t)(i + 1) * KC, K, BN, KC, BN, K);
        __builtin_amdgcn_s_wait_tensorcnt(2);   // chunk i landed, i+1 in flight
      } else {
        __builtin_amdgcn_s_wait_tensorcnt(0);
      }
    }
    __syncthreads();                       // buf i visible to all waves
    const bfr* Ar0 = &Asm[buf][(w * 32 + nl) * KC];
    const bfr* Ar1 = Ar0 + 16 * KC;
    const bfr* Bs = &Bsm[buf][nl * KC];
#pragma unroll
    for (int kloc = 0; kloc < KC; kloc += 32) {
      v16bf a0 = ldA(Ar0, kloc, kh);       // ds_load_b128 x2
      v16bf a1 = ldA(Ar1, kloc, kh);
      v16bf b0 = ldB(Bs + 0 * 16 * KC, kloc, kh);
      v16bf b1 = ldB(Bs + 1 * 16 * KC, kloc, kh);
      v16bf b2 = ldB(Bs + 2 * 16 * KC, kloc, kh);
      v16bf b3 = ldB(Bs + 3 * 16 * KC, kloc, kh);
      acc[0][0] = wmma_bf16(a0, b0, acc[0][0]);
      acc[1][0] = wmma_bf16(a1, b0, acc[1][0]);
      acc[0][1] = wmma_bf16(a0, b1, acc[0][1]);
      acc[1][1] = wmma_bf16(a1, b1, acc[1][1]);
      acc[0][2] = wmma_bf16(a0, b2, acc[0][2]);
      acc[1][2] = wmma_bf16(a1, b2, acc[1][2]);
      acc[0][3] = wmma_bf16(a0, b3, acc[0][3]);
      acc[1][3] = wmma_bf16(a1, b3, acc[1][3]);
    }
    __syncthreads();                       // done reading before TDM overwrites
  }

  int col0 = bn * BN;
  int mb = bm * BM + w * 32;
#pragma unroll
  for (int ct = 0; ct < 4; ct++) {
    int col = col0 + ct * 16 + nl;
    float bv = bias ? bias[col] : 0.f;
#pragma unroll
    for (int g = 0; g < 2; g++) {
#pragma unroll
      for (int r = 0; r < 8; r++) {
        int mrow = mb + g * 16 + r + kh * 8;
        float v = acc[g][ct][r] + bv;
        if (relu) v = fmaxf(v, 0.f);
        if (Cf) Cf[(size_t)mrow * Nc + col] = v;
        if (Cb) Cb[(size_t)mrow * Nc + col] = f2bf(v);
      }
    }
  }
}

// ---------------- fused edge kernel (TDM-staged) ------------------------------
// One block (16 waves, 512 threads) per 16-edge tile. Wave 0 TDM-loads the
// 16x512 bf16 edge tile into LDS once; wave w = (h, dq) computes four WMMA
// chains (Ew0,Ew1,Eb0,Eb1) over K=512 with A from LDS, then gathers
// K[src]+Q[dst], signed-sqrt, relu; emits e_out bf16 + reduced logits.
__global__ __launch_bounds__(512) void k_edge(
    const bfr* __restrict__ Ein, const bfr* __restrict__ WEt,   // E x 512, 1024 x 512
    const float* __restrict__ bE, const float* __restrict__ Aw, // 1024, H*D
    const float* __restrict__ Qm, const float* __restrict__ Km, // N x 512 f32
    const int* __restrict__ src, const int* __restrict__ dst,
    bfr* __restrict__ Eout, float* __restrict__ logits, int E) {
  __shared__ __align__(16) bfr Etile[16 * DIMV];
  int w = threadIdx.x >> 5;                      // 0..15
  int lane = threadIdx.x & 31;
  int et = blockIdx.x;                           // edge tile
  int h = w >> 2, dq = w & 3;                    // head, 32-col quarter
  int kh = lane >> 4, nl = lane & 15;
  if (w == 0) {                                  // wave-uniform TDM issue
    tdm_load_2d((unsigned)(uintptr_t)&Etile[0], Ein + (size_t)et * 16 * DIMV,
                DIMV, 16, DIMV, 16, DIMV);
    __builtin_amdgcn_s_wait_tensorcnt(0);
  }
  __syncthreads();                               // tile visible to all waves
  const bfr* Arow = &Etile[nl * DIMV];
  int cb = h * 256 + dq * 32;                    // col base in 1024-wide EWB
  const bfr* Bw0 = WEt + (size_t)(cb + nl) * DIMV;
  const bfr* Bw1 = WEt + (size_t)(cb + 16 + nl) * DIMV;
  const bfr* Bb0 = WEt + (size_t)(cb + 128 + nl) * DIMV;
  const bfr* Bb1 = WEt + (size_t)(cb + 144 + nl) * DIMV;
  v8f aw0, aw1, ab0, ab1;
#pragma unroll
  for (int r = 0; r < 8; r++) { aw0[r] = 0.f; aw1[r] = 0.f; ab0[r] = 0.f; ab1[r] = 0.f; }
#pragma unroll 2
  for (int k0 = 0; k0 < DIMV; k0 += 32) {
    v16bf a = ldA(Arow, k0, kh);                 // ds_load_b128 x2
    v16bf b0 = ldB(Bw0, k0, kh);
    v16bf b1 = ldB(Bw1, k0, kh);
    v16bf b2 = ldB(Bb0, k0, kh);
    v16bf b3 = ldB(Bb1, k0, kh);
    aw0 = wmma_bf16(a, b0, aw0);
    aw1 = wmma_bf16(a, b1, aw1);
    ab0 = wmma_bf16(a, b2, ab0);
    ab1 = wmma_bf16(a, b3, ab1);
  }
  int d0 = h * DHEAD + dq * 32 + nl, d1 = d0 + 16;
  float bw0 = bE[cb + nl], bw1 = bE[cb + 16 + nl];
  float bb0 = bE[cb + 128 + nl], bb1 = bE[cb + 144 + nl];
  float awv0 = Aw[d0], awv1 = Aw[d1];
#pragma unroll
  for (int r = 0; r < 8; r++) {
    int e = et * 16 + r + kh * 8;
    int sn = src[e], dn = dst[e];
    float kq0 = Km[(size_t)sn * DIMV + d0] + Qm[(size_t)dn * DIMV + d0];
    float kq1 = Km[(size_t)sn * DIMV + d1] + Qm[(size_t)dn * DIMV + d1];
    float sw0 = kq0 * (aw0[r] + bw0);
    float sw1 = kq1 * (aw1[r] + bw1);
    float ss0 = sqrtf(fmaxf(sw0, 0.f)) - sqrtf(fmaxf(-sw0, 0.f));
    float ss1 = sqrtf(fmaxf(sw1, 0.f)) - sqrtf(fmaxf(-sw1, 0.f));
    float sv0 = fmaxf(ss0 + ab0[r] + bb0, 0.f);
    float sv1 = fmaxf(ss1 + ab1[r] + bb1, 0.f);
    if (Eout) {
      Eout[(size_t)e * DIMV + d0] = f2bf(sv0);
      Eout[(size_t)e * DIMV + d1] = f2bf(sv1);
    }
    float part = sv0 * awv0 + sv1 * awv1;        // partial logit for this lane
#pragma unroll
    for (int off = 1; off < 16; off <<= 1) part += __shfl_xor(part, off, 32);
    if (nl == 0) atomicAdd(&logits[(size_t)e * NHEAD + h], part);
  }
}

// ---------------- scatter softmax ----------------
__device__ __forceinline__ unsigned ord_enc(float f) {
  unsigned u = __float_as_uint(f);
  return (u & 0x80000000u) ? ~u : (u | 0x80000000u);
}
__device__ __forceinline__ float ord_dec(unsigned e) {
  unsigned u = (e & 0x80000000u) ? (e & 0x7FFFFFFFu) : ~e;
  return __uint_as_float(u);
}
__global__ void k_segmax(const float* __restrict__ logits, const int* __restrict__ dst,
                         unsigned* __restrict__ menc, int E) {
  int t = blockIdx.x * blockDim.x + threadIdx.x;
  if (t >= E * NHEAD) return;
  int e = t >> 2, h = t & 3;
  atomicMax(&menc[(size_t)dst[e] * NHEAD + h], ord_enc(logits[t]));
}
__global__ void k_segexp(const float* __restrict__ logits, const int* __restrict__ dst,
                         const unsigned* __restrict__ menc, float* __restrict__ wb,
                         float* __restrict__ z, int E) {
  int t = blockIdx.x * blockDim.x + threadIdx.x;
  if (t >= E * NHEAD) return;
  int e = t >> 2, h = t & 3;
  int dn = dst[e];
  float w = __expf(logits[t] - ord_dec(menc[(size_t)dn * NHEAD + h]));
  wb[t] = w;
  atomicAdd(&z[(size_t)dn * NHEAD + h], w);
}
__global__ void k_msg(const float* __restrict__ V, const int* __restrict__ src,
                      const int* __restrict__ dst, const float* __restrict__ wb,
                      const float* __restrict__ z, float* __restrict__ outx, int E) {
  int t = blockIdx.x * blockDim.x + threadIdx.x;
  if (t >= E * (DIMV / 4)) return;
  int e = t >> 7, c4 = t & 127;
  int c = c4 * 4, h = c >> 7;
  int dn = dst[e], sn = src[e];
  float alpha = wb[(size_t)e * NHEAD + h] / (z[(size_t)dn * NHEAD + h] + 1e-16f);
  const float4 v = *reinterpret_cast<const float4*>(V + (size_t)sn * DIMV + c);
  float* o = outx + (size_t)dn * DIMV + c;
  atomicAdd(o + 0, v.x * alpha);
  atomicAdd(o + 1, v.y * alpha);
  atomicAdd(o + 2, v.z * alpha);
  atomicAdd(o + 3, v.w * alpha);
}

// ---------------- misc elementwise ----------------
__global__ void k_x0(const float* __restrict__ query, const float* __restrict__ npos,
                     bfr* __restrict__ xb, int pos) {
  int t = blockIdx.x * blockDim.x + threadIdx.x;
  if (t >= N_NODES * DIMV) return;
  int n = t >> 9, c = t & 511;
  xb[t] = f2bf(query[(size_t)n * SEQ * DIMV + (size_t)pos * DIMV + c] + npos[t]);
}
__global__ void k_f2b(const float* __restrict__ X, bfr* __restrict__ Y, int n) {
  int t = blockIdx.x * blockDim.x + threadIdx.x;
  if (t < n) Y[t] = f2bf(X[t]);
}
__global__ void k_gather(const float* __restrict__ outx, const int* __restrict__ map,
                         float* __restrict__ dout, int pos) {
  int t = blockIdx.x * blockDim.x + threadIdx.x;
  if (t >= BOUT * DIMV) return;
  int b = t >> 9, c = t & 511;
  dout[(size_t)b * SEQ * DIMV + (size_t)pos * DIMV + c] =
      outx[(size_t)map[b] * DIMV + c];
}

// ---------------- host orchestration ----------------
static inline long cdiv(long a, long b) { return (a + b - 1) / b; }

extern "C" void kernel_launch(void* const* d_in, const int* in_sizes, int n_in,
                              void* d_out, int out_size, void* d_ws, size_t ws_size,
                              hipStream_t stream) {
  (void)in_sizes; (void)n_in; (void)out_size; (void)ws_size;
  const float* query = (const float*)d_in[0];
  const int*   eidx  = (const int*)d_in[1];
  const int*   srcI  = eidx;
  const int*   dstI  = eidx + E_TOT;
  const int*   mapI  = (const int*)d_in[2];
  const float* emb   = (const float*)d_in[3];
  const float* wn_w1 = (const float*)d_in[5];
  const float* wn_b1 = (const float*)d_in[6];
  const float* wn_w2 = (const float*)d_in[7];
  const float* wn_b2 = (const float*)d_in[8];
  const float* wp_w1 = (const float*)d_in[9];
  const float* wp_b1 = (const float*)d_in[10];
  const float* wp_w2 = (const float*)d_in[11];
  const float* wp_b2 = (const float*)d_in[12];
  const float* WQ    = (const float*)d_in[13];
  const float* WK    = (const float*)d_in[14];
  const float* WV    = (const float*)d_in[15];
  const float* WE    = (const float*)d_in[16];
  const float* bE    = (const float*)d_in[17];
  const float* Aw    = (const float*)d_in[18];
  float* dout = (float*)d_out;

  char* ws = (char*)d_ws;
  size_t off = 0;
  auto carve = [&](size_t bytes) -> void* {
    void* p = ws + off;
    off = (off + bytes + 255) & ~(size_t)255;
    return p;
  };
  bfr* wn_w2t = (bfr*)carve((size_t)DIMV * DIMV * 2);
  bfr* wp_w2t = (bfr*)carve((size_t)DIMV * DIMV * 2);
  bfr* WQt = (bfr*)carve((size_t)NLAY * DIMV * DIMV * 2);
  bfr* WKt = (bfr*)carve((size_t)NLAY * DIMV * DIMV * 2);
  bfr* WVt = (bfr*)carve((size_t)NLAY * DIMV * DIMV * 2);
  bfr* WEt = (bfr*)carve((size_t)NLAY * 1024 * DIMV * 2);
  float* nraw = (float*)carve((size_t)N_NODES * 8 * 4);
  bfr* h1 = (bfr*)carve((size_t)N_NODES * DIMV * 2);
  float* npos = (float*)carve((size_t)N_NODES * DIMV * 4);
  bfr* e0 = (bfr*)carve((size_t)E_TOT * DIMV * 2);
  bfr* e1 = (bfr*)carve((size_t)E_TOT * DIMV * 2);  // also MLP hidden 'he'
  bfr* xb = (bfr*)carve((size_t)N_NODES * DIMV * 2);
  float* Qm = (float*)carve((size_t)N_NODES * DIMV * 4);
  float* Km = (float*)carve((size_t)N_NODES * DIMV * 4);
  float* Vm = (float*)carve((size_t)N_NODES * DIMV * 4);
  float* logits = (float*)carve((size_t)E_TOT * NHEAD * 4);
  unsigned* menc = (unsigned*)carve((size_t)N_NODES * NHEAD * 4);
  float* zb = (float*)carve((size_t)N_NODES * NHEAD * 4);
  float* wb = (float*)carve((size_t)E_TOT * NHEAD * 4);
  float* outx = (float*)carve((size_t)N_NODES * DIMV * 4);

  const int T = 256;

  // ---- weight convert + transpose ----
  {
    long n = (long)DIMV * DIMV;
    k_wconv<<<cdiv(n, T), T, 0, stream>>>(wn_w2, wn_w2t, DIMV, DIMV);
    k_wconv<<<cdiv(n, T), T, 0, stream>>>(wp_w2, wp_w2t, DIMV, DIMV);
    for (int l = 0; l < NLAY; l++) {
      k_wconv<<<cdiv(n, T), T, 0, stream>>>(WQ + (size_t)l * n, WQt + (size_t)l * n, DIMV, DIMV);
      k_wconv<<<cdiv(n, T), T, 0, stream>>>(WK + (size_t)l * n, WKt + (size_t)l * n, DIMV, DIMV);
      k_wconv<<<cdiv(n, T), T, 0, stream>>>(WV + (size_t)l * n, WVt + (size_t)l * n, DIMV, DIMV);
      long ne = (long)DIMV * 1024;
      k_wconv<<<cdiv(ne, T), T, 0, stream>>>(WE + (size_t)l * ne, WEt + (size_t)l * ne, DIMV, 1024);
    }
  }

  // ---- node positional features ----
  hipMemsetAsync(nraw, 0, (size_t)N_NODES * 8 * 4, stream);
  k_selfloop<<<cdiv(E_TOT, T), T, 0, stream>>>(srcI, dstI, emb, nraw, E_TOT);
  k_mlp1<<<cdiv((long)N_NODES * DIMV, T), T, 0, stream>>>(nraw, wn_w1, wn_b1, h1, N_NODES);
  k_gemm<<<(N_NODES / BM) * (DIMV / BN), T, 0, stream>>>(h1, wn_w2t, wn_b2, npos,
                                                         nullptr, N_NODES, DIMV, DIMV, 0);
  // ---- edge positional features e0 ----
  k_mlp1<<<cdiv((long)E_TOT * DIMV, T), T, 0, stream>>>(emb, wp_w1, wp_b1, e1 /*he*/, E_TOT);
  k_gemm<<<(E_TOT / BM) * (DIMV / BN), T, 0, stream>>>(e1 /*he*/, wp_w2t, wp_b2,
                                                       nullptr, e0, E_TOT, DIMV, DIMV, 0);

  const long gblocks = (long)(N_NODES / BM) * (DIMV / BN);

  // ---- per query position ----
  for (int p = 0; p < SEQ; p++) {
    k_x0<<<cdiv((long)N_NODES * DIMV, T), T, 0, stream>>>(query, npos, xb, p);
    for (int l = 0; l < NLAY; l++) {
      size_t wo = (size_t)l * DIMV * DIMV;
      k_gemm<<<gblocks, T, 0, stream>>>(xb, WQt + wo, nullptr, Qm, nullptr,
                                        N_NODES, DIMV, DIMV, 0);
      k_gemm<<<gblocks, T, 0, stream>>>(xb, WKt + wo, nullptr, Km, nullptr,
                                        N_NODES, DIMV, DIMV, 0);
      k_gemm<<<gblocks, T, 0, stream>>>(xb, WVt + wo, nullptr, Vm, nullptr,
                                        N_NODES, DIMV, DIMV, 0);
      hipMemsetAsync(logits, 0, (size_t)E_TOT * NHEAD * 4, stream);
      hipMemsetAsync(menc, 0, (size_t)N_NODES * NHEAD * 4, stream);
      hipMemsetAsync(zb, 0, (size_t)N_NODES * NHEAD * 4, stream);
      hipMemsetAsync(outx, 0, (size_t)N_NODES * DIMV * 4, stream);
      const bfr* ein = (l == 0) ? e0 : e1;
      bfr* eout = (l == 0) ? e1 : nullptr;  // last layer's e_out is dead
      k_edge<<<E_TOT / 16, 512, 0, stream>>>(
          ein, WEt + (size_t)l * 1024 * DIMV, bE + (size_t)l * 1024,
          Aw + (size_t)l * NHEAD * DHEAD, Qm, Km, srcI, dstI, eout, logits, E_TOT);
      k_segmax<<<cdiv((long)E_TOT * NHEAD, T), T, 0, stream>>>(logits, dstI, menc, E_TOT);
      k_segexp<<<cdiv((long)E_TOT * NHEAD, T), T, 0, stream>>>(logits, dstI, menc, wb, zb, E_TOT);
      k_msg<<<cdiv((long)E_TOT * (DIMV / 4), T), T, 0, stream>>>(Vm, srcI, dstI, wb, zb,
                                                                 outx, E_TOT);
      if (l + 1 < NLAY)
        k_f2b<<<cdiv((long)N_NODES * DIMV, T), T, 0, stream>>>(outx, xb, N_NODES * DIMV);
    }
    k_gather<<<cdiv((long)BOUT * DIMV, T), T, 0, stream>>>(outx, mapI, dout, p);
  }
}
